// LongTermAttention_18640158064770
// MI455X (gfx1250) — matrix operand order repaired and checked by reading, not compiled, not run on hardware
//
#include <hip/hip_runtime.h>
#include <hip/hip_bf16.h>

typedef __attribute__((ext_vector_type(16))) __bf16         v16bf;
typedef __attribute__((ext_vector_type(8)))  float          v8f;
typedef __attribute__((ext_vector_type(8)))  unsigned short us8;
typedef __attribute__((ext_vector_type(4)))  int            v4i;

#define BMT 128   // block tile M (wave strip = 32 rows)
#define BNT 64    // block tile N
#define BKT 32    // K step (one WMMA depth)

#if defined(__has_builtin)
#if __has_builtin(__builtin_amdgcn_global_load_async_to_lds_b128) && \
    __has_builtin(__builtin_amdgcn_s_wait_asynccnt)
#define USE_ASYNC_LDS 1
#endif
#endif

union FragBF { v16bf v; unsigned short u[16]; us8 h[2]; };

__device__ __forceinline__ unsigned short f2bf(float f) {
  unsigned int x = __float_as_uint(f);
  unsigned int r = (x + 0x7FFFu + ((x >> 16) & 1u)) >> 16;   // RNE
  return (unsigned short)r;
}
__device__ __forceinline__ float bf2f(unsigned short u) {
  return __uint_as_float(((unsigned int)u) << 16);
}

#if USE_ASYNC_LDS
__device__ __forceinline__ void async_ld16(const void* g, void* lds) {
  __builtin_amdgcn_global_load_async_to_lds_b128(
      (__attribute__((address_space(1))) v4i*)(unsigned long long)(g),
      (__attribute__((address_space(3))) v4i*)(lds), 0, 0);
}
#endif

// Vectorized f32 -> bf16 (n must be a multiple of 8; n8 = n/8)
__global__ __launch_bounds__(256)
void f32_to_bf16_v(const float* __restrict__ in, unsigned short* __restrict__ out,
                   size_t n8) {
  size_t i = (size_t)blockIdx.x * blockDim.x + threadIdx.x;
  size_t st = (size_t)gridDim.x * blockDim.x;
  for (; i < n8; i += st) {
    const float4* p = (const float4*)(in + i * 8);
    float4 x = p[0], y = p[1];
    us8 o;
    o[0] = f2bf(x.x); o[1] = f2bf(x.y); o[2] = f2bf(x.z); o[3] = f2bf(x.w);
    o[4] = f2bf(y.x); o[5] = f2bf(y.y); o[6] = f2bf(y.z); o[7] = f2bf(y.w);
    *(us8*)(out + i * 8) = o;
  }
}

// Tiled transpose + convert: in[z][R,C] f32 -> out[z][C,R] bf16. R,C multiples of 64.
__global__ __launch_bounds__(256)
void transpose_f32_to_bf16(const float* __restrict__ in, unsigned short* __restrict__ out,
                           int R, int C, long long ibs, long long obs) {
  __shared__ unsigned short t[64][65];
  in  += (long long)blockIdx.z * ibs;
  out += (long long)blockIdx.z * obs;
  const int r0 = blockIdx.y * 64, c0 = blockIdx.x * 64;
  for (int i = threadIdx.x; i < 64 * 64; i += 256) {
    int rr = i >> 6, cc = i & 63;
    t[rr][cc] = f2bf(in[(long long)(r0 + rr) * C + (c0 + cc)]);
  }
  __syncthreads();
  for (int i = threadIdx.x; i < 64 * 64; i += 256) {
    int cc = i >> 6, rr = i & 63;
    out[(long long)(c0 + cc) * R + (r0 + rr)] = t[rr][cc];
  }
}

// Batched GEMM: C[z][m,n] = alpha * sum_k A[z][m,k]*B[z][k,n].
// K-contiguous operands REQUIRED (sAk == sBk == 1). A,B bf16; C bf16 or f32.
// 128 threads = 4 wave32; block tile 128x64x32, each wave owns a 32x64 strip
// (2 A-frags x 4 B-frags = 8 WMMA per K-step, each B-frag reused twice).
// Double-buffered LDS; async global->LDS staging when available.
__global__ __launch_bounds__(128)
void gemm_bf16_wmma(const unsigned short* __restrict__ Ag,
                    const unsigned short* __restrict__ Bg,
                    void* __restrict__ Cg,
                    int K,
                    long long sAm, long long sBn, long long sCm, long long sCn,
                    long long bsA, long long hsA,
                    long long bsB, long long hsB,
                    long long bsC, long long hsC,
                    int nh, float alpha, int out_f32)
{
  __shared__ __align__(16) unsigned short As[2][BMT * BKT];   // 16 KB
  __shared__ __align__(16) unsigned short Bs[2][BNT * BKT];   //  8 KB

  const int tid  = threadIdx.x;
  const int wave = tid >> 5;
  const int lane = tid & 31;
  const int mrow = lane & 15;
  const int sel  = lane >> 4;

  const int z  = blockIdx.z;
  const int zb = z / nh;
  const int zh = z - zb * nh;
  Ag += zb * bsA + zh * hsA;
  Bg += zb * bsB + zh * hsB;

  const long long tm = (long long)blockIdx.y * BMT;
  const long long tn = (long long)blockIdx.x * BNT;

  // A tile: 128 rows x 4 16B-chunks = 512 chunks -> 4 per thread.
  // B tile:  64 rows x 4 16B-chunks = 256 chunks -> 2 per thread.
  const int crow = tid >> 2;           // 0..31
  const int ckof = (tid & 3) * 8;      // 0,8,16,24
  const unsigned short* pA[4];
  int soA[4];
#pragma unroll
  for (int t = 0; t < 4; ++t) {
    pA[t]  = Ag + (tm + crow + 32 * t) * sAm + ckof;
    soA[t] = (crow + 32 * t) * BKT + ckof;
  }
  const unsigned short* pB0 = Bg + (tn + crow) * sBn + ckof;
  const unsigned short* pB1 = Bg + (tn + crow + 32) * sBn + ckof;
  const int soB0 = crow * BKT + ckof;
  const int soB1 = soB0 + 32 * BKT;

  // prologue: stage tile 0 into buffer 0
#if USE_ASYNC_LDS
#pragma unroll
  for (int t = 0; t < 4; ++t) async_ld16(pA[t], &As[0][soA[t]]);
  async_ld16(pB0, &Bs[0][soB0]);
  async_ld16(pB1, &Bs[0][soB1]);
  __builtin_amdgcn_s_wait_asynccnt(0);
#else
#pragma unroll
  for (int t = 0; t < 4; ++t) *(us8*)&As[0][soA[t]] = *(const us8*)pA[t];
  *(us8*)&Bs[0][soB0] = *(const us8*)pB0;
  *(us8*)&Bs[0][soB1] = *(const us8*)pB1;
#endif
  __syncthreads();

  v8f acc[2][4] = {};
  int buf = 0;
  for (int kk = 0; kk < K; kk += BKT) {
    const int  nxt  = buf ^ 1;
    const bool more = (kk + BKT) < K;

#if USE_ASYNC_LDS
    if (more) {
#pragma unroll
      for (int t = 0; t < 4; ++t) async_ld16(pA[t] + kk + BKT, &As[nxt][soA[t]]);
      async_ld16(pB0 + kk + BKT, &Bs[nxt][soB0]);
      async_ld16(pB1 + kk + BKT, &Bs[nxt][soB1]);
    }
#else
    us8 va[4], vb0, vb1;
    if (more) {
#pragma unroll
      for (int t = 0; t < 4; ++t) va[t] = *(const us8*)(pA[t] + kk + BKT);
      vb0 = *(const us8*)(pB0 + kk + BKT);
      vb1 = *(const us8*)(pB1 + kk + BKT);
      if (kk + 2 * BKT < K) {
        __builtin_prefetch(pA[0] + kk + 2 * BKT, 0, 1);
        __builtin_prefetch(pB0 + kk + 2 * BKT, 0, 1);
      }
    }
#endif

    // Two A fragments (16x32 bf16 each, ISA layout): 2x two 16B LDS reads
    FragBF a[2];
#pragma unroll
    for (int i = 0; i < 2; ++i) {
      const unsigned short* arow = &As[buf][(wave * 32 + i * 16 + mrow) * BKT];
      a[i].h[0] = *(const us8*)(arow + sel * 8);
      a[i].h[1] = *(const us8*)(arow + 16 + sel * 8);
    }

#pragma unroll
    for (int j = 0; j < 4; ++j) {
      FragBF b;
      const unsigned short* brow = &Bs[buf][(j * 16 + mrow) * BKT + sel * 8];
      b.h[0] = *(const us8*)(brow);
      b.h[1] = *(const us8*)(brow + 16);
      acc[0][j] = __builtin_amdgcn_wmma_f32_16x16x32_bf16(
          false, a[0].v, false, b.v, (short)0, acc[0][j], false, false);
      acc[1][j] = __builtin_amdgcn_wmma_f32_16x16x32_bf16(
          false, a[1].v, false, b.v, (short)0, acc[1][j], false, false);
    }

#if USE_ASYNC_LDS
    if (more) __builtin_amdgcn_s_wait_asynccnt(0);
#else
    if (more) {
#pragma unroll
      for (int t = 0; t < 4; ++t) *(us8*)&As[nxt][soA[t]] = va[t];
      *(us8*)&Bs[nxt][soB0] = vb0;
      *(us8*)&Bs[nxt][soB1] = vb1;
    }
#endif
    __syncthreads();
    buf = nxt;
  }

  // C/D layout: VGPR r -> M = sel*8 + r, N = lane&15
  const long long cbase = zb * bsC + zh * hsC;
#pragma unroll
  for (int i = 0; i < 2; ++i) {
#pragma unroll
    for (int j = 0; j < 4; ++j) {
      long long col = tn + j * 16 + mrow;
#pragma unroll
      for (int r = 0; r < 8; ++r) {
        long long row = tm + wave * 32 + i * 16 + sel * 8 + r;
        float vv = acc[i][j][r] * alpha;
        if (out_f32) ((float*)Cg)[cbase + row * sCm + col * sCn] = vv;
        else ((unsigned short*)Cg)[cbase + row * sCm + col * sCn] = f2bf(vv);
      }
    }
  }
}

// One wave32 per (b,h,q) row of scores[.,512]; lane owns 16 contiguous elems.
// Dots with w_mu/w_sigma -> sigmoid/softplus -> overwrite row with Gaussian r.
__global__ __launch_bounds__(256)
void mu_sigma_r_k(unsigned short* __restrict__ sc,
                  const float* __restrict__ w_mu,
                  const float* __restrict__ w_sigma,
                  const float* __restrict__ basis_mu,
                  const float* __restrict__ basis_sigma,
                  long long rows)
{
  long long row = (long long)blockIdx.x * 8 + (threadIdx.x >> 5);
  if (row >= rows) return;
  const int lane = threadIdx.x & 31;
  const int base = lane * 16;
  unsigned short* rp = sc + row * 512 + base;

  us8 s0 = ((const us8*)rp)[0];
  us8 s1 = ((const us8*)rp)[1];
  float sv[16];
#pragma unroll
  for (int j = 0; j < 8; ++j) { sv[j] = bf2f(s0[j]); sv[8 + j] = bf2f(s1[j]); }

  float dm = 0.f, dsg = 0.f;
#pragma unroll
  for (int j = 0; j < 16; ++j) {
    dm  += sv[j] * w_mu[base + j];
    dsg += sv[j] * w_sigma[base + j];
  }
#pragma unroll
  for (int off = 16; off >= 1; off >>= 1) {   // wave32 butterfly
    dm  += __shfl_xor(dm, off, 32);
    dsg += __shfl_xor(dsg, off, 32);
  }
  float mu   = 1.f / (1.f + __expf(-dm));
  float sp   = (dsg > 20.f) ? dsg : log1pf(__expf(dsg));
  float sig2 = fmaxf(sp, 1e-4f);

  us8 o0, o1;
#pragma unroll
  for (int j = 0; j < 16; ++j) {
    float bs  = basis_sigma[base + j];
    float var = sig2 + bs * bs;
    float d   = mu - basis_mu[base + j];
    float rr  = __expf(-0.5f * d * d / var) * rsqrtf(6.283185307179586f * var);
    unsigned short hb = f2bf(rr);
    if (j < 8) o0[j] = hb; else o1[j - 8] = hb;
  }
  ((us8*)rp)[0] = o0;
  ((us8*)rp)[1] = o1;
}

extern "C" void kernel_launch(void* const* d_in, const int* in_sizes, int n_in,
                              void* d_out, int out_size, void* d_ws, size_t ws_size,
                              hipStream_t stream) {
  const int Bc = 4, Lc = 2048, Qc = 2048, Hc = 16, Dc = 64, Ec = 1024, NBc = 512;
  (void)in_sizes; (void)n_in; (void)out_size; (void)ws_size;

  const float* k_f  = (const float*)d_in[0];
  const float* q_f  = (const float*)d_in[1];
  const float* Wq_f = (const float*)d_in[2];
  const float* Wk_f = (const float*)d_in[3];
  const float* Wv_f = (const float*)d_in[4];
  const float* Wo_f = (const float*)d_in[5];
  const float* wmu  = (const float*)d_in[6];
  const float* wsg  = (const float*)d_in[7];
  const float* Gs_f = (const float*)d_in[8];
  const float* bmu  = (const float*)d_in[9];
  const float* bsg  = (const float*)d_in[10];

  char* w = (char*)d_ws;
  size_t off = 0;
  auto carve = [&](size_t elems) -> unsigned short* {
    unsigned short* p = (unsigned short*)(w + off);
    off += ((elems * 2 + 255) / 256) * 256;
    return p;
  };
  unsigned short* kTb  = carve((size_t)Bc * Ec * Lc);        // k^T  [B,E,L]
  unsigned short* qb   = carve((size_t)Bc * Qc * Ec);
  unsigned short* Wqb  = carve((size_t)Ec * Ec);
  unsigned short* Wkb  = carve((size_t)Ec * Ec);
  unsigned short* Wvb  = carve((size_t)Ec * Ec);
  unsigned short* Wob  = carve((size_t)Ec * Ec);
  unsigned short* Gstb = carve((size_t)NBc * Lc);            // Gs^T [NB,L]
  unsigned short* Bmb  = carve((size_t)Bc * NBc * Ec);
  unsigned short* keyb = carve((size_t)Bc * NBc * Ec);
  unsigned short* valTb= carve((size_t)Bc * Ec * NBc);       // vals^T [B,E,NB]
  unsigned short* qryb = carve((size_t)Bc * Qc * Ec);
  unsigned short* scb  = carve((size_t)Bc * Hc * Qc * NBc);  // scores -> r in place
  unsigned short* ctxb = carve((size_t)Bc * Qc * Ec);

  // conversions / transposes
  f32_to_bf16_v<<<1024, 256, 0, stream>>>(q_f,  qb,  (size_t)Bc * Qc * Ec / 8);
  f32_to_bf16_v<<<256,  256, 0, stream>>>(Wq_f, Wqb, (size_t)Ec * Ec / 8);
  f32_to_bf16_v<<<256,  256, 0, stream>>>(Wk_f, Wkb, (size_t)Ec * Ec / 8);
  f32_to_bf16_v<<<256,  256, 0, stream>>>(Wv_f, Wvb, (size_t)Ec * Ec / 8);
  f32_to_bf16_v<<<256,  256, 0, stream>>>(Wo_f, Wob, (size_t)Ec * Ec / 8);
  transpose_f32_to_bf16<<<dim3(NBc / 64, Lc / 64, 1), 256, 0, stream>>>(
      Gs_f, Gstb, Lc, NBc, 0, 0);
  transpose_f32_to_bf16<<<dim3(Ec / 64, Lc / 64, Bc), 256, 0, stream>>>(
      k_f, kTb, Lc, Ec, (long long)Lc * Ec, (long long)Ec * Lc);

  // G1: Bm[b] = Gs^T @ k[b]   (M=NB,N=E,K=L)  A=Gst[n,l], B=kT[e,l]
  gemm_bf16_wmma<<<dim3(Ec / 64, NBc / 128, Bc), 128, 0, stream>>>(
      Gstb, kTb, Bmb, Lc,
      Lc, Lc, Ec, 1,
      0, 0, (long long)Ec * Lc, 0, (long long)NBc * Ec, 0,
      1, 1.0f, 0);

  // G2: keys[b] = Bm[b] @ Wk^T   (M=NB,N=E,K=E)
  gemm_bf16_wmma<<<dim3(Ec / 64, NBc / 128, Bc), 128, 0, stream>>>(
      Bmb, Wkb, keyb, Ec,
      Ec, Ec, Ec, 1,
      (long long)NBc * Ec, 0, 0, 0, (long long)NBc * Ec, 0,
      1, 1.0f, 0);

  // G3: vals^T: C[m=nb, n=e'] stored at valT[e'*NB + nb]  (sCm=1, sCn=NB)
  gemm_bf16_wmma<<<dim3(Ec / 64, NBc / 128, Bc), 128, 0, stream>>>(
      Bmb, Wvb, valTb, Ec,
      Ec, Ec, 1, NBc,
      (long long)NBc * Ec, 0, 0, 0, (long long)Ec * NBc, 0,
      1, 1.0f, 0);

  // G4: qry[b] = q[b] @ Wq^T   (M=Q,N=E,K=E)
  gemm_bf16_wmma<<<dim3(Ec / 64, Qc / 128, Bc), 128, 0, stream>>>(
      qb, Wqb, qryb, Ec,
      Ec, Ec, Ec, 1,
      (long long)Qc * Ec, 0, 0, 0, (long long)Qc * Ec, 0,
      1, 1.0f, 0);

  // G5: scores[b,h] = qry.h @ keys.h^T / 8   (M=Q,N=NB,K=D), z = B*H
  gemm_bf16_wmma<<<dim3(NBc / 64, Qc / 128, Bc * Hc), 128, 0, stream>>>(
      qryb, keyb, scb, Dc,
      Ec, Ec, NBc, 1,
      (long long)Qc * Ec, Dc, (long long)NBc * Ec, Dc,
      (long long)Hc * Qc * NBc, (long long)Qc * NBc,
      Hc, 0.125f, 0);

  // G6: scores -> r (in place)
  long long rows = (long long)Bc * Hc * Qc;
  mu_sigma_r_k<<<(unsigned)(rows / 8), 256, 0, stream>>>(scb, wmu, wsg, bmu, bsg, rows);

  // G7: ctx[b,h] = r @ vals[b,h]   (M=Q,N=D,K=NB); B=valT[e',nb]
  gemm_bf16_wmma<<<dim3(Dc / 64, Qc / 128, Bc * Hc), 128, 0, stream>>>(
      scb, valTb, ctxb, NBc,
      NBc, NBc, Ec, 1,
      (long long)Hc * Qc * NBc, (long long)Qc * NBc,
      (long long)Ec * NBc, (long long)Dc * NBc,
      (long long)Qc * Ec, Dc,
      Hc, 1.0f, 0);

  // G8: out = ctx @ Wo^T   (M=B*Q,N=E,K=E) -> f32 d_out
  gemm_bf16_wmma<<<dim3(Ec / 64, (Bc * Qc) / 128, 1), 128, 0, stream>>>(
      ctxb, Wob, d_out, Ec,
      Ec, Ec, Ec, 1,
      0, 0, 0, 0, 0, 0,
      1, 1.0f, 1);
}
